// SelfAttention_54520314855519
// MI455X (gfx1250) — matrix operand order repaired and checked by reading, not compiled, not run on hardware
//
#include <hip/hip_runtime.h>

// ---------------------------------------------------------------------------
// Multi-head self-attention for MI455X (gfx1250), flash-attention fused,
// f16 WMMA (v_wmma_f32_16x16x32_f16) with fp32 accumulation.
// K/V blocks (64-wide) staged to LDS with gfx1250 async-to-LDS copies,
// double-buffered.   B=2, S=2048, E=1024, H=16, D=64
// ---------------------------------------------------------------------------

typedef __attribute__((ext_vector_type(16))) _Float16 v16h;
typedef __attribute__((ext_vector_type(8)))  _Float16 v8h;
typedef __attribute__((ext_vector_type(8)))  float    v8f;
typedef __attribute__((ext_vector_type(4)))  float    v4f;

#define WMMA(a, b, c) \
  __builtin_amdgcn_wmma_f32_16x16x32_f16(false, (a), false, (b), (short)0, (c), false, false)

static constexpr int S_LEN = 2048;
static constexpr int EMB   = 1024;
static constexpr int HEADS = 16;
static constexpr int HD    = 64;                 // head dim
static constexpr size_t ELEMS = (size_t)2 * S_LEN * EMB;  // 4,194,304 per buffer

// ---- fragment loaders -------------------------------------------------------
// A fragment (16x32, f16): lane holds row (lane&15); halves 0..7 = K[k0+hi*8 ..],
// halves 8..15 = K[k0+16+hi*8 ..]  (two contiguous 8-half runs).
__device__ inline v16h load_a_f16(const _Float16* rowp, int k0, int hi) {
  v8h lo8 = *(const v8h*)(rowp + k0 + hi * 8);
  v8h hi8 = *(const v8h*)(rowp + k0 + 16 + hi * 8);
  v16h a;
#pragma unroll
  for (int i = 0; i < 8; ++i) { a[i] = lo8[i]; a[8 + i] = hi8[i]; }
  return a;
}

__device__ inline v16h load_a_f32(const float* __restrict__ rowp, int k0, int hi) {
  const float* p0 = rowp + k0 + hi * 8;
  const float* p1 = p0 + 16;
  v4f x0 = *(const v4f*)(p0);
  v4f x1 = *(const v4f*)(p0 + 4);
  v4f x2 = *(const v4f*)(p1);
  v4f x3 = *(const v4f*)(p1 + 4);
  v16h a;
#pragma unroll
  for (int i = 0; i < 4; ++i) {
    a[i]      = (_Float16)x0[i];
    a[4 + i]  = (_Float16)x1[i];
    a[8 + i]  = (_Float16)x2[i];
    a[12 + i] = (_Float16)x3[i];
  }
  return a;
}

// B fragment (32x16, f16): lane holds col n=(lane&15); halves 0..7 = K[k0+hi*8 ..],
// halves 8..15 = K[k0+16+hi*8 ..] of the "B-source row" (row-major [n][k] source).
__device__ inline v16h load_b_f16(const _Float16* rowp, int k0, int hi) {
  v8h lo8 = *(const v8h*)(rowp + k0 + hi * 8);
  v8h hi8 = *(const v8h*)(rowp + k0 + 16 + hi * 8);
  v16h b;
#pragma unroll
  for (int i = 0; i < 8; ++i) { b[i] = lo8[i]; b[8 + i] = hi8[i]; }
  return b;
}

__device__ inline v16h load_b_f32(const float* __restrict__ rowp, int k0, int hi) {
  const float* p0 = rowp + k0 + hi * 8;
  const float* p1 = p0 + 16;
  v4f x0 = *(const v4f*)(p0);
  v4f x1 = *(const v4f*)(p0 + 4);
  v4f x2 = *(const v4f*)(p1);
  v4f x3 = *(const v4f*)(p1 + 4);
  v16h b;
#pragma unroll
  for (int i = 0; i < 4; ++i) {
    b[i]      = (_Float16)x0[i];
    b[4 + i]  = (_Float16)x1[i];
    b[8 + i]  = (_Float16)x2[i];
    b[12 + i] = (_Float16)x3[i];
  }
  return b;
}

// gfx1250 async copy: 16 bytes per lane, global -> LDS, tracked by ASYNCcnt.
__device__ inline void async_copy_b128(unsigned lds_byte_addr, const void* gptr) {
  unsigned long long ga = (unsigned long long)gptr;
  asm volatile("global_load_async_to_lds_b128 %0, %1, off"
               :: "v"(lds_byte_addr), "v"(ga) : "memory");
}
__device__ inline void wait_asynccnt0() {
  asm volatile("s_wait_asynccnt 0x0" ::: "memory");
}

// ---------------------------------------------------------------------------
// Kernel 1: QKV projection.  x viewed as [B*S*H, 64] rows (layout-exact).
// q scaled by 1/sqrt(E)=1/32 at store.  Q,K stored [B,H,S,D] f16; V stored
// transposed [B,H,D,S] f16 so P*V B-fragments are contiguous.
// ---------------------------------------------------------------------------
__global__ __launch_bounds__(256) void qkv_proj(
    const float* __restrict__ x,
    const float* __restrict__ Wq, const float* __restrict__ Wk, const float* __restrict__ Wv,
    _Float16* __restrict__ Qh, _Float16* __restrict__ Kh, _Float16* __restrict__ VTh) {
  const int lane = threadIdx.x & 31;
  const int w    = threadIdx.x >> 5;
  const int lo   = lane & 15;
  const int hi   = lane >> 4;
  const int mbase = blockIdx.x * 128 + w * 16;

  const float* xrow = x + (size_t)(mbase + lo) * HD;
  v16h a0 = load_a_f32(xrow, 0, hi);
  v16h a1 = load_a_f32(xrow, 32, hi);

  const float* Ws[3] = {Wq, Wk, Wv};
#pragma unroll
  for (int m = 0; m < 3; ++m) {
    const float* W = Ws[m];
#pragma unroll
    for (int nt = 0; nt < 4; ++nt) {
      // B[k][n] = W[n][k]; lane's source row = W row (nt*16+lo), contiguous in k.
      const float* wrow = W + (size_t)(nt * 16 + lo) * HD;
      v8f c = {};
      c = WMMA(a0, load_b_f32(wrow, 0, hi), c);
      c = WMMA(a1, load_b_f32(wrow, 32, hi), c);
      // convert first, then store (avoid per-store data-reg reuse stalls)
      _Float16 hv[8];
      size_t ofs[8];
#pragma unroll
      for (int r = 0; r < 8; ++r) {
        const int mrow = mbase + r + 8 * hi;           // flat row in [B*S*H]
        const int h = mrow & 15;
        const int s = (mrow >> 4) & (S_LEN - 1);
        const int b = mrow >> 15;                      // / (S*H)
        const int col = nt * 16 + lo;
        const float v = c[r];
        if (m == 0) {
          hv[r]  = (_Float16)(v * 0.03125f);
          ofs[r] = ((size_t)(b * HEADS + h) * S_LEN + s) * HD + col;
        } else if (m == 1) {
          hv[r]  = (_Float16)v;
          ofs[r] = ((size_t)(b * HEADS + h) * S_LEN + s) * HD + col;
        } else {
          hv[r]  = (_Float16)v;
          ofs[r] = ((size_t)(b * HEADS + h) * HD + col) * S_LEN + s;
        }
      }
      _Float16* dst = (m == 0) ? Qh : (m == 1) ? Kh : VTh;
#pragma unroll
      for (int r = 0; r < 8; ++r) dst[ofs[r]] = hv[r];
    }
  }
}

// ---------------------------------------------------------------------------
// Kernel 2: flash attention per (b,h).  Each wave owns 16 q rows; online
// softmax over k-blocks of 64.  K (64x64) and V^T (64x64) blocks are staged
// into LDS with async-to-LDS copies (double buffered, 4 async ops per wave
// per block) shared by all 8 waves.  P tile (16x64) goes through a per-wave
// LDS slab to convert C/D layout -> A layout.
//
// LDS map (halves): buf0 K[4096] V[4096] | buf1 K[4096] V[4096] | P 8*1024
// ---------------------------------------------------------------------------
__global__ __launch_bounds__(256) void flash_attn(
    const _Float16* __restrict__ Qh, const _Float16* __restrict__ Kh,
    const _Float16* __restrict__ VTh, _Float16* __restrict__ attn) {
  __shared__ __align__(32) _Float16 smem[2 * 8192 + 8 * 1024];  // 48 KB

  const int tid  = threadIdx.x;
  const int lane = tid & 31;
  const int w    = tid >> 5;
  const int lo   = lane & 15;
  const int hi   = lane >> 4;
  const int bh   = blockIdx.y;                      // b*16 + h
  const int qbase = blockIdx.x * 128 + w * 16;

  const _Float16* Qb = Qh + (size_t)bh * S_LEN * HD;
  const _Float16* Kb = Kh + (size_t)bh * S_LEN * HD;
  const _Float16* Vb = VTh + (size_t)bh * HD * S_LEN;
  _Float16* pw = smem + 2 * 8192 + w * 1024;        // [16][64] halves

  const unsigned smem_base = (unsigned)(unsigned long long)(void*)smem;

  const _Float16* qrow = Qb + (size_t)(qbase + lo) * HD;
  const v16h q0 = load_a_f16(qrow, 0, hi);
  const v16h q1 = load_a_f16(qrow, 32, hi);

  float mrow[8], lrow[8];
  v8f acc[4];
#pragma unroll
  for (int r = 0; r < 8; ++r) { mrow[r] = -INFINITY; lrow[r] = 0.0f; }
#pragma unroll
  for (int t = 0; t < 4; ++t) acc[t] = (v8f){};

  // Stage one 64-k block (K: contiguous 8KB span; V^T: 64 rows x 128B).
  // 512 x 16B chunks each; thread t owns chunks 2t, 2t+1.
  auto stage = [&](int buf, int kb) {
    const unsigned kbase = smem_base + (unsigned)(buf * 8192) * 2u;
    const unsigned vbase = kbase + 4096u * 2u;
#pragma unroll
    for (int j = 0; j < 2; ++j) {
      const int c = tid * 2 + j;
      async_copy_b128(kbase + (unsigned)c * 16u, Kb + (size_t)kb * HD + c * 8);
      async_copy_b128(vbase + (unsigned)c * 16u,
                      Vb + (size_t)(c >> 3) * S_LEN + kb + (c & 7) * 8);
    }
  };

  stage(0, 0);

  const int NBLK = S_LEN / 64;                      // 32 blocks
  for (int i = 0; i < NBLK; ++i) {
    const int cur = i & 1;
    wait_asynccnt0();          // this wave's stage of block i complete
    __syncthreads();           // all waves' stages visible in LDS

    if (i + 1 < NBLK) stage(cur ^ 1, (i + 1) * 64);

    const _Float16* Kblk = smem + cur * 8192;         // [64][64] halves
    const _Float16* Vblk = smem + cur * 8192 + 4096;  // [64][64] halves

    // --- scores: 16x64 block, four 16x16 tiles (from LDS) ------------------
    v8f st[4];
#pragma unroll
    for (int t = 0; t < 4; ++t) {
      const _Float16* krow = Kblk + (size_t)(t * 16 + lo) * HD;  // col n = k index
      v8f c = {};
      c = WMMA(q0, load_b_f16(krow, 0, hi), c);
      c = WMMA(q1, load_b_f16(krow, 32, hi), c);
      st[t] = c;
    }
    // --- online softmax over 64 columns ------------------------------------
#pragma unroll
    for (int r = 0; r < 8; ++r) {
      float mx = fmaxf(fmaxf(st[0][r], st[1][r]), fmaxf(st[2][r], st[3][r]));
      mx = fmaxf(mx, __shfl_xor(mx, 1, 32));
      mx = fmaxf(mx, __shfl_xor(mx, 2, 32));
      mx = fmaxf(mx, __shfl_xor(mx, 4, 32));
      mx = fmaxf(mx, __shfl_xor(mx, 8, 32));
      const float mnew = fmaxf(mrow[r], mx);
      float p[4];
#pragma unroll
      for (int t = 0; t < 4; ++t) p[t] = __expf(st[t][r] - mnew);
      float sum = (p[0] + p[1]) + (p[2] + p[3]);
      sum += __shfl_xor(sum, 1, 32);
      sum += __shfl_xor(sum, 2, 32);
      sum += __shfl_xor(sum, 4, 32);
      sum += __shfl_xor(sum, 8, 32);
      const float scale = __expf(mrow[r] - mnew);
      lrow[r] = lrow[r] * scale + sum;
      mrow[r] = mnew;
#pragma unroll
      for (int t = 0; t < 4; ++t) acc[t][r] *= scale;
      const int row = r + 8 * hi;                    // C/D layout row
#pragma unroll
      for (int t = 0; t < 4; ++t)
        pw[row * 64 + t * 16 + lo] = (_Float16)p[t];
    }
    asm volatile("s_wait_dscnt 0" ::: "memory");
    // --- P (A layout, 16x64 = two 16x32 frags) x V (from LDS) --------------
    const v16h pa0 = load_a_f16(pw + lo * 64, 0, hi);
    const v16h pa1 = load_a_f16(pw + lo * 64, 32, hi);
#pragma unroll
    for (int t = 0; t < 4; ++t) {
      const _Float16* vrow = Vblk + (size_t)(t * 16 + lo) * 64;  // V^T row = d
      acc[t] = WMMA(pa0, load_b_f16(vrow, 0, hi), acc[t]);
      acc[t] = WMMA(pa1, load_b_f16(vrow, 32, hi), acc[t]);
    }
  }

  // --- epilogue: attn[(b*S+s)][h*64+d] as f16 -------------------------------
  const int b = bh >> 4, h = bh & 15;
#pragma unroll
  for (int r = 0; r < 8; ++r) {
    const float inv = 1.0f / lrow[r];
    const int qi = qbase + r + 8 * hi;
    const size_t rowoff = ((size_t)b * S_LEN + qi) * EMB + h * HD;
#pragma unroll
    for (int t = 0; t < 4; ++t)
      attn[rowoff + t * 16 + lo] = (_Float16)(acc[t][r] * inv);
  }
}

// ---------------------------------------------------------------------------
// Kernel 3: out = attn @ W_out^T + b_out.  M=4096, N=K=1024.
// Each wave computes a 16x64 fp32 tile; W_out converted to f16 on load.
// ---------------------------------------------------------------------------
__global__ __launch_bounds__(256) void out_proj(
    const _Float16* __restrict__ attn, const float* __restrict__ Wout,
    const float* __restrict__ bout, float* __restrict__ out) {
  const int lane = threadIdx.x & 31;
  const int w    = threadIdx.x >> 5;
  const int lo   = lane & 15;
  const int hi   = lane >> 4;
  const int wid  = blockIdx.x * 8 + w;
  const int mt    = wid >> 4;            // 0..255 -> 16-row tiles
  const int nbase = (wid & 15) * 64;

  const _Float16* arow = attn + (size_t)(mt * 16 + lo) * EMB;
  v8f acc[4];
#pragma unroll
  for (int t = 0; t < 4; ++t) acc[t] = (v8f){};

  for (int k0 = 0; k0 < EMB; k0 += 32) {
    const v16h a = load_a_f16(arow, k0, hi);
#pragma unroll
    for (int t = 0; t < 4; ++t) {
      const float* wrow = Wout + (size_t)(nbase + t * 16 + lo) * EMB;  // B[k][n]=Wout[n][k]
      acc[t] = WMMA(a, load_b_f32(wrow, k0, hi), acc[t]);
    }
  }

#pragma unroll
  for (int t = 0; t < 4; ++t) {
    const int col = nbase + t * 16 + lo;
    const float bb = bout[col];
#pragma unroll
    for (int r = 0; r < 8; ++r) {
      const int row = mt * 16 + r + 8 * hi;
      out[(size_t)row * EMB + col] = acc[t][r] + bb;
    }
  }
}

// ---------------------------------------------------------------------------
extern "C" void kernel_launch(void* const* d_in, const int* in_sizes, int n_in,
                              void* d_out, int out_size, void* d_ws, size_t ws_size,
                              hipStream_t stream) {
  const float* x    = (const float*)d_in[0];
  const float* Wq   = (const float*)d_in[1];
  const float* Wk   = (const float*)d_in[2];
  const float* Wv   = (const float*)d_in[3];
  const float* Wout = (const float*)d_in[4];
  const float* bout = (const float*)d_in[5];
  float* out = (float*)d_out;

  _Float16* ws   = (_Float16*)d_ws;
  _Float16* Qh   = ws;                 // [B,H,S,D] f16 (pre-scaled by 1/32)
  _Float16* Kh   = ws + ELEMS;         // [B,H,S,D] f16
  _Float16* VTh  = ws + 2 * ELEMS;     // [B,H,D,S] f16
  _Float16* attn = ws + 3 * ELEMS;     // [B*S, E]  f16

  // 65536 rows / 128 rows per block
  qkv_proj<<<512, 256, 0, stream>>>(x, Wq, Wk, Wv, Qh, Kh, VTh);
  // grid: (S/128 q-blocks, B*H heads)
  flash_attn<<<dim3(16, 32), 256, 0, stream>>>(Qh, Kh, VTh, attn);
  // 4096x1024 output, 8 wave-tiles (16x64) per block
  out_proj<<<512, 256, 0, stream>>>(attn, Wout, bout, out);
}
